// Attention_15882789061200
// MI455X (gfx1250) — matrix operand order repaired and checked by reading, not compiled, not run on hardware
//
#include <hip/hip_runtime.h>

// ---------------------------------------------------------------------------
// Fused windowed attention for MI455X (gfx1250, wave32, WMMA 16x16x32 f16).
//   B=96 (b*gx*gy), N=320 (l*w1*w2), D=256, H=8, DHEAD=32.
// ---------------------------------------------------------------------------

typedef _Float16 v16h __attribute__((ext_vector_type(16)));
typedef _Float16 v8h  __attribute__((ext_vector_type(8)));
typedef _Float16 v4h  __attribute__((ext_vector_type(4)));
typedef float    v8f  __attribute__((ext_vector_type(8)));
typedef float    v4f  __attribute__((ext_vector_type(4)));

// Workspace layout (all _Float16 elements)
static constexpr long OW_QKVT = 0;                          // [768][256]  wq|wk|wv transposed
static constexpr long OW_OUTT = OW_QKVT + 768L * 256;       // [32][256]   w_out transposed
static constexpr long OW_Q    = OW_OUTT + 32L * 256;        // [96][8][320][32]
static constexpr long OW_K    = OW_Q  + 96L * 8 * 320 * 32; // [96][8][320][32]
static constexpr long OW_VT   = OW_K  + 96L * 8 * 320 * 32; // [96][8][32][320]  (V transposed)
static constexpr long OW_O    = OW_VT + 96L * 8 * 320 * 32; // [96][320][256]    attn out, ch = h*32+d

__device__ __forceinline__ v16h cat8(v8h a, v8h b) {
  v16h r;
#pragma unroll
  for (int i = 0; i < 8; ++i) { r[i] = a[i]; r[i + 8] = b[i]; }
  return r;
}

__device__ __forceinline__ v8f wmma_f16(v16h a, v16h b, v8f c) {
  // v_wmma_f32_16x16x32_f16  (neg_a, A, neg_b, B, c_mod, C, reuse_a, reuse_b)
  return __builtin_amdgcn_wmma_f32_16x16x32_f16(false, a, false, b, (short)0, c,
                                                false, false);
}

// ---------------------------------------------------------------------------
// Kernel 0: convert + transpose weights to f16 so WMMA B-fragments are
// contiguous per lane.  wqkvT[n][k]; woutT[j][k].
// ---------------------------------------------------------------------------
__global__ void __launch_bounds__(256) prep_weights_kernel(
    const float* __restrict__ wq, const float* __restrict__ wkv,
    const float* __restrict__ wout, _Float16* __restrict__ ws) {
  const int i = blockIdx.x * blockDim.x + threadIdx.x;
  const int NW1 = 768 * 256;
  if (i < NW1) {
    const int n = i >> 8, k = i & 255;
    const float v = (n < 256) ? wq[k * 256 + n] : wkv[k * 512 + (n - 256)];
    ws[OW_QKVT + (long)n * 256 + k] = (_Float16)v;
  } else if (i < NW1 + 32 * 256) {
    const int j = i - NW1;
    const int row = j >> 8, c = j & 255;
    ws[OW_OUTT + (long)row * 256 + c] = (_Float16)wout[c * 32 + row];
  }
}

// ---------------------------------------------------------------------------
// Kernel 1: LayerNorm + QKV projection.
// grid (1920, 6) x 256 threads (8 waves).  Each block: one 16-row tile of the
// re-gathered xs[(b gx gy)][(l w1 w2)][256], LN into LDS f16, then each wave
// computes one 16x16 tile of the 768-wide QKV projection with 8 WMMAs.
// ---------------------------------------------------------------------------
__global__ void __launch_bounds__(256) ln_qkv_kernel(
    const float* __restrict__ x, const float* __restrict__ lng,
    const float* __restrict__ lnb, _Float16* __restrict__ ws) {
  __shared__ _Float16 xs[16][264];      // padded: 264*2 = 528B row stride
  __shared__ float redS[16][16];
  __shared__ float redQ[16][16];

  const int rt   = blockIdx.x;          // 0..1919 : 16-row tile over 96*320 rows
  const int bidx = rt / 20;             // batch  (b*gx*gy)
  const int n0   = (rt % 20) * 16;      // token base
  const int t    = threadIdx.x;
  const int r    = t >> 4, seg = t & 15;

  // gather permutation: xs[bidx][n] <- x[b][l][gx][gy][w1][w2][:]
  {
    const int n  = n0 + r;
    const int bb = bidx / 48, gx = (bidx % 48) >> 3, gy = bidx & 7;
    const int l  = n / 64, w1 = (n & 63) >> 3, w2 = n & 7;
    const long xoff =
        ((((((long)bb * 5 + l) * 6 + gx) * 8 + gy) * 8 + w1) * 8 + w2) * 256;
    float vals[16];
    float s = 0.f, s2 = 0.f;
#pragma unroll
    for (int i = 0; i < 16; ++i) {
      const float v = x[xoff + seg * 16 + i];
      vals[i] = v; s += v; s2 += v * v;
    }
    redS[r][seg] = s; redQ[r][seg] = s2;
    __syncthreads();
    float ts = 0.f, tq = 0.f;
#pragma unroll
    for (int i = 0; i < 16; ++i) { ts += redS[r][i]; tq += redQ[r][i]; }
    const float mu   = ts * (1.0f / 256.0f);
    const float var  = tq * (1.0f / 256.0f) - mu * mu;
    const float rstd = rsqrtf(var + 1e-5f);
#pragma unroll
    for (int i = 0; i < 16; ++i) {
      const int c = seg * 16 + i;
      xs[r][c] = (_Float16)((vals[i] - mu) * rstd * lng[c] + lnb[c]);
    }
    __syncthreads();
  }

  // WMMA: one 16x16 tile per wave, K = 256 (8 steps of 32)
  const int wave = t >> 5, lane = t & 31;
  const int lcol = lane & 15, sel = lane >> 4;
  const int j0   = (blockIdx.y * 8 + wave) * 16;   // output column base, < 768
  const _Float16* wT = ws + OW_QKVT + (long)(j0 + lcol) * 256;

  v8f acc = {};
#pragma unroll
  for (int kk = 0; kk < 256; kk += 32) {
    const v8h lo = *(const v8h*)&xs[lcol][kk + sel * 16];
    const v8h hi = *(const v8h*)&xs[lcol][kk + sel * 16 + 8];
    const v16h a = cat8(lo, hi);
    const v16h b = *(const v16h*)(wT + kk + sel * 16);
    acc = wmma_f16(a, b, acc);
  }

  // Scatter: j<256 -> Q[b][h][n][d]; j<512 -> K; else V stored transposed.
  const int j = j0 + lcol;              // fixed per lane
#pragma unroll
  for (int v = 0; v < 8; ++v) {
    const int  gn  = n0 + v + 8 * sel;
    const float vf = acc[v];
    if (j < 256) {
      const int h = j >> 5, d = j & 31;
      ws[OW_Q + (((long)bidx * 8 + h) * 320 + gn) * 32 + d] = (_Float16)vf;
    } else if (j < 512) {
      const int jj = j - 256, h = jj >> 5, d = jj & 31;
      ws[OW_K + (((long)bidx * 8 + h) * 320 + gn) * 32 + d] = (_Float16)vf;
    } else {
      const int jj = j - 512, h = jj >> 5, d = jj & 31;
      ws[OW_VT + (((long)bidx * 8 + h) * 32 + d) * 320 + gn] = (_Float16)vf;
    }
  }
}

// ---------------------------------------------------------------------------
// Kernel 2: attention per (batch, head).  768 blocks x 128 threads (4 waves).
// Each wave owns 5 query tiles; scores+bias live in a per-wave LDS row buffer,
// softmax fully vectorized (float4 LDS passes, wave32 __shfl_xor(16) reduce),
// P -> f16 LDS (v4h stores), then P.V via WMMA.
// Dynamic LDS: 4*16*320*(4+2) = 122880 bytes.
// ---------------------------------------------------------------------------
__global__ void __launch_bounds__(128) attn_kernel(
    const float* __restrict__ context, _Float16* __restrict__ ws) {
  extern __shared__ char smem[];
  float*    scores = (float*)smem;                                     // [4][16][320]
  _Float16* Pbuf   = (_Float16*)(smem + 4 * 16 * 320 * sizeof(float)); // [4][16][320]

  const int bh   = blockIdx.x;          // 0..767
  const int bidx = bh >> 3, h = bh & 7;
  const int t    = threadIdx.x, wave = t >> 5, lane = t & 31;
  const int lcol = lane & 15, sel = lane >> 4;

  float*    myS = scores + (long)wave * 16 * 320;
  _Float16* myP = Pbuf   + (long)wave * 16 * 320;
  const _Float16* Qb  = ws + OW_Q  + (long)bh * 320 * 32;
  const _Float16* Kb  = ws + OW_K  + (long)bh * 320 * 32;
  const _Float16* Vtb = ws + OW_VT + (long)bh * 32 * 320;
  const float*    ctxb = context + (long)bidx * 320 * 2560 + h * 320;

  for (int qt = wave; qt < 20; qt += 4) {     // exactly 5 iters per wave
    const int n0 = qt * 16;

    // prefetch the bias rows this tile will consume
#pragma unroll
    for (int v = 0; v < 8; ++v) {
      __builtin_prefetch(ctxb + (long)(n0 + v + 8 * sel) * 2560 + lcol * 16, 0, 0);
    }

    // Q A-fragment (16x32 f16)
    const _Float16* qr = Qb + (long)(n0 + lcol) * 32;
    const v16h aq = cat8(*(const v8h*)(qr + sel * 16),
                         *(const v8h*)(qr + sel * 16 + 8));

    // S = Q K^T + bias  -> LDS row buffer (bias rows read 64B-coalesced)
    for (int mt = 0; mt < 20; ++mt) {
      const int m0 = mt * 16;
      const v16h bk = *(const v16h*)(Kb + (long)(m0 + lcol) * 32 + sel * 16);
      v8f s = {};
      s = wmma_f16(aq, bk, s);
      const int m = m0 + lcol;
#pragma unroll
      for (int v = 0; v < 8; ++v) {
        const int rloc = v + 8 * sel;
        const float bias = ctxb[(long)(n0 + rloc) * 2560 + m];
        myS[rloc * 320 + m] = s[v] + bias;
      }
    }
    __syncthreads();

    // softmax over 320 cols: lane pair (lcol, lcol+16) splits the row.
    // All passes vectorized: ds_load/store_b128 on the score row, v4h P out.
    {
      v4f* rowv = (v4f*)(myS + lcol * 320 + sel * 160);   // 40 x float4
      // pass 1: max
      v4f mx4 = {-3.0e38f, -3.0e38f, -3.0e38f, -3.0e38f};
      for (int i = 0; i < 40; ++i) {
        const v4f v = rowv[i];
#pragma unroll
        for (int j = 0; j < 4; ++j) mx4[j] = fmaxf(mx4[j], v[j]);
      }
      float mx = fmaxf(fmaxf(mx4[0], mx4[1]), fmaxf(mx4[2], mx4[3]));
      mx = fmaxf(mx, __shfl_xor(mx, 16, 32));
      // pass 2: exp + sum (exponentials kept in LDS)
      v4f sum4 = {};
      for (int i = 0; i < 40; ++i) {
        v4f v = rowv[i];
#pragma unroll
        for (int j = 0; j < 4; ++j) v[j] = __expf(v[j] - mx);
        rowv[i] = v;
        sum4 += v;
      }
      float sum = (sum4[0] + sum4[1]) + (sum4[2] + sum4[3]);
      sum += __shfl_xor(sum, 16, 32);
      const float inv = 1.0f / sum;
      // pass 3: normalize -> f16 P
      v4h* prow = (v4h*)(myP + lcol * 320 + sel * 160);
      for (int i = 0; i < 40; ++i) {
        const v4f v = rowv[i];
        v4h p;
#pragma unroll
        for (int j = 0; j < 4; ++j) p[j] = (_Float16)(v[j] * inv);
        prow[i] = p;
      }
    }
    __syncthreads();

    // O = P (16x320) * V (320x32): 10 K-steps, two 16-wide output halves
    v8f acc0 = {}, acc1 = {};
    for (int mk = 0; mk < 320; mk += 32) {
      const _Float16* pr = myP + lcol * 320 + mk;
      const v16h ap = cat8(*(const v8h*)(pr + sel * 16),
                           *(const v8h*)(pr + sel * 16 + 8));
      const v16h b0 = *(const v16h*)(Vtb + (long)lcol * 320 + mk + sel * 16);
      const v16h b1 = *(const v16h*)(Vtb + (long)(16 + lcol) * 320 + mk + sel * 16);
      acc0 = wmma_f16(ap, b0, acc0);
      acc1 = wmma_f16(ap, b1, acc1);
    }
#pragma unroll
    for (int v = 0; v < 8; ++v) {
      const int rloc = v + 8 * sel;
      const long orow = ((long)bidx * 320 + n0 + rloc) * 256 + h * 32;
      ws[OW_O + orow + lcol]      = (_Float16)acc0[v];
      ws[OW_O + orow + 16 + lcol] = (_Float16)acc1[v];
    }
    __syncthreads();
  }
}

// ---------------------------------------------------------------------------
// Kernel 3: out = O @ w_out  (30720 x 256 x 32) + final un-window scatter.
// 960 blocks x 128 threads (4 waves): wave -> (row-tile half, col-tile).
// ---------------------------------------------------------------------------
__global__ void __launch_bounds__(128) outproj_kernel(
    const _Float16* __restrict__ ws, float* __restrict__ out) {
  const int t = threadIdx.x, wave = t >> 5, lane = t & 31;
  const int lcol = lane & 15, sel = lane >> 4;
  const int rt2 = blockIdx.x * 2 + (wave >> 1);   // 0..1919
  const int jt  = wave & 1;                       // d-head column half

  const _Float16* Ar = ws + OW_O + (long)(rt2 * 16 + lcol) * 256;
  const _Float16* Wo = ws + OW_OUTT + (long)(jt * 16 + lcol) * 256;

  v8f acc = {};
#pragma unroll
  for (int kk = 0; kk < 256; kk += 32) {
    const v16h a = cat8(*(const v8h*)(Ar + kk + sel * 16),
                        *(const v8h*)(Ar + kk + sel * 16 + 8));
    const v16h b = *(const v16h*)(Wo + kk + sel * 16);
    acc = wmma_f16(a, b, acc);
  }

#pragma unroll
  for (int v = 0; v < 8; ++v) {
    const int grow = rt2 * 16 + v + 8 * sel;      // 0..30719
    const int bidx = grow / 320, n = grow % 320;
    const int bb = bidx / 48, gx = (bidx % 48) >> 3, gy = bidx & 7;
    const int l = n / 64, w1 = (n & 63) >> 3, w2 = n & 7;
    const int j = jt * 16 + lcol;
    const long o =
        ((((((long)bb * 5 + l) * 6 + gx) * 8 + gy) * 8 + w1) * 8 + w2) * 32 + j;
    out[o] = acc[v];
  }
}

// ---------------------------------------------------------------------------
extern "C" void kernel_launch(void* const* d_in, const int* in_sizes, int n_in,
                              void* d_out, int out_size, void* d_ws, size_t ws_size,
                              hipStream_t stream) {
  (void)in_sizes; (void)n_in; (void)out_size; (void)ws_size;
  const float* x    = (const float*)d_in[0];
  const float* ctx  = (const float*)d_in[1];
  const float* wq   = (const float*)d_in[2];
  const float* wkv  = (const float*)d_in[3];
  const float* wout = (const float*)d_in[4];
  const float* lng  = (const float*)d_in[5];
  const float* lnb  = (const float*)d_in[6];
  _Float16* ws = (_Float16*)d_ws;
  float* out = (float*)d_out;

  const int prep_elems = 768 * 256 + 32 * 256;
  prep_weights_kernel<<<(prep_elems + 255) / 256, 256, 0, stream>>>(wq, wkv, wout, ws);

  dim3 g1(1920, 6);
  ln_qkv_kernel<<<g1, 256, 0, stream>>>(x, lng, lnb, ws);

  attn_kernel<<<768, 128, 122880, stream>>>(ctx, ws);

  outproj_kernel<<<960, 128, 0, stream>>>(ws, out);
}
